// HybridLayer_57784490001038
// MI455X (gfx1250) — compile-verified, matrix operand order checked by reading, not run on hardware
//
#include <hip/hip_runtime.h>
#include <hip/hip_bf16.h>

// ---------------- types & helpers ----------------
typedef __attribute__((ext_vector_type(16))) __bf16          v16bf;
typedef __attribute__((ext_vector_type(8)))  float           v8f;
typedef __attribute__((ext_vector_type(8)))  unsigned        v8u;

#if defined(__gfx1250__) && __has_builtin(__builtin_amdgcn_cvt_pk_bf16_f32)
__device__ __forceinline__ unsigned pack2(float lo, float hi) {
    return __builtin_bit_cast(unsigned, __builtin_amdgcn_cvt_pk_bf16_f32(lo, hi));
}
#else
// native fptrunc: selects the hardware bf16 convert on gfx1250 when available
__device__ __forceinline__ unsigned pack2(float lo, float hi) {
    unsigned short a = __builtin_bit_cast(unsigned short, (__bf16)lo);
    unsigned short b = __builtin_bit_cast(unsigned short, (__bf16)hi);
    return (unsigned)a | ((unsigned)b << 16);
}
#endif
__device__ __forceinline__ v8f wmma_bf16(v16bf a, v16bf b, v8f c) {
    return __builtin_amdgcn_wmma_f32_16x16x32_bf16(false, a, false, b, (short)0, c, false, false);
}
// packed-u32 word p of a 16x32 bf16 A/B fragment holds K = kpat(p,hi), kpat(p,hi)+1
__device__ __forceinline__ int kpat(int p, int hi) { return ((p < 4) ? 2 * p : 2 * p + 8) + 8 * hi; }

#define D_   768
#define L_   1024
#define B_   8
#define H_   12
#define HD_  64

// -------- CDNA5 async global->LDS copy (ASYNCcnt path), with safe fallback --------
#if defined(__gfx1250__) && __has_builtin(__builtin_amdgcn_global_load_async_to_lds_b32)
#define ASYNC_LDS 1
typedef __attribute__((address_space(1))) int gas_int;   // global
typedef __attribute__((address_space(3))) int las_int;   // LDS
__device__ __forceinline__ void async_cp_f32(const float* g, float* l) {
    __builtin_amdgcn_global_load_async_to_lds_b32((gas_int*)g, (las_int*)l, 0, 0);
}
__device__ __forceinline__ void async_wait0() {
#if __has_builtin(__builtin_amdgcn_s_wait_asynccnt)
    __builtin_amdgcn_s_wait_asynccnt(0);
#else
    asm volatile("s_wait_asynccnt 0" ::: "memory");
#endif
}
#endif

// ---------------- generic bf16 WMMA GEMM:  C = op(A[MxK] @ W[KxN] (+C) (+bias)) ----------------
// Block tile 64(M) x 128(N), 8 waves; each wave: one 16-row A fragment x four 16-col B tiles.
// Double-buffered LDS staging; OP/ACC are compile-time.
template <int OP, int ACC>
__global__ __launch_bounds__(256) void gemm_bf16(const float* __restrict__ A,
                                                 const float* __restrict__ W,
                                                 const float* __restrict__ bias,
                                                 float* __restrict__ C,
                                                 int M, int N, int K) {
    __shared__ unsigned As32[2][64 * 17];    // 64 rows x 16 packed-k u32 (pad 17)
    __shared__ unsigned Bs32[2][128 * 17];   // 128 cols x 16 packed-k u32
    const int m0 = blockIdx.y * 64, n0 = blockIdx.x * 128;
    const int tid = threadIdx.x, wid = tid >> 5, lane = tid & 31;
    const int lanelo = lane & 15, hi = lane >> 4;
    const int wm = wid >> 1, wn = wid & 1;

    v8f c[4];
    #pragma unroll
    for (int t = 0; t < 4; ++t) c[t] = (v8f){0,0,0,0,0,0,0,0};

    unsigned ra[4], rb[8];
    auto loadA = [&](int k0) {
        #pragma unroll
        for (int j = 0; j < 4; ++j) {
            int i = tid + j * 256, row = i >> 4, kp = i & 15;
            const float* ap = A + (long)(m0 + row) * K + k0 + 2 * kp;
            float2 v = *(const float2*)ap;
            ra[j] = pack2(v.x, v.y);
        }
        if (k0 + 64 < K)   // speculative prefetch of the K-tile two steps ahead
            __builtin_prefetch(A + (long)(m0 + (tid >> 4)) * K + k0 + 64, 0, 0);
    };
    auto loadB = [&](int k0) {
        #pragma unroll
        for (int j = 0; j < 8; ++j) {
            int i = tid + j * 256, n = i & 127, kp = i >> 7;
            const float* wp = W + (long)(k0 + 2 * kp) * N + n0 + n;
            rb[j] = pack2(wp[0], wp[N]);
        }
    };
    auto storeA = [&](int buf) {
        #pragma unroll
        for (int j = 0; j < 4; ++j) {
            int i = tid + j * 256, row = i >> 4, kp = i & 15;
            As32[buf][row * 17 + kp] = ra[j];
        }
    };
    auto storeB = [&](int buf) {
        #pragma unroll
        for (int j = 0; j < 8; ++j) {
            int i = tid + j * 256, n = i & 127, kp = i >> 7;
            Bs32[buf][n * 17 + kp] = rb[j];
        }
    };

    loadA(0); loadB(0);
    storeA(0); storeB(0);
    __syncthreads();

    const int nk = K >> 5;
    for (int kk = 0; kk < nk; ++kk) {
        const int cur = kk & 1, nxt = cur ^ 1;
        if (kk + 1 < nk) { loadA((kk + 1) * 32); loadB((kk + 1) * 32); }

        const int arow = wm * 16 + lanelo;
        v8u af;
        #pragma unroll
        for (int p = 0; p < 8; ++p)
            af[p] = As32[cur][arow * 17 + (kpat(p, hi) >> 1)];
        v16bf aF = __builtin_bit_cast(v16bf, af);
        #pragma unroll
        for (int t = 0; t < 4; ++t) {
            int bcol = wn * 64 + t * 16 + lanelo;
            v8u bfv;
            #pragma unroll
            for (int p = 0; p < 8; ++p)
                bfv[p] = Bs32[cur][bcol * 17 + (kpat(p, hi) >> 1)];
            c[t] = wmma_bf16(aF, __builtin_bit_cast(v16bf, bfv), c[t]);
        }
        if (kk + 1 < nk) { storeA(nxt); storeB(nxt); }
        __syncthreads();
    }

    #pragma unroll
    for (int t = 0; t < 4; ++t) {
        int n = n0 + wn * 64 + t * 16 + lanelo;
        float bv = bias ? bias[n] : 0.f;
        #pragma unroll
        for (int r = 0; r < 8; ++r) {
            int m = m0 + wm * 16 + r + 8 * hi;
            long idx = (long)m * N + n;
            float v = c[t][r];
            if constexpr (ACC) v += C[idx];
            v += bv;
            if constexpr (OP == 1) v = fmaxf(v, 0.f);
            else if constexpr (OP == 2) v = 1.f / (1.f + __expf(-v));
            else if constexpr (OP == 3) v = tanhf(v);
            C[idx] = v;
        }
    }
}

static void launch_gemm(const float* A, const float* W, const float* bias, float* C,
                        int M, int N, int K, int op, int acc, hipStream_t s) {
    dim3 g(N / 128, M / 64);
    if (acc == 0) {
        if (op == 2) gemm_bf16<2, 0><<<g, 256, 0, s>>>(A, W, bias, C, M, N, K);
        else         gemm_bf16<0, 0><<<g, 256, 0, s>>>(A, W, bias, C, M, N, K);
    } else {
        if (op == 2) gemm_bf16<2, 1><<<g, 256, 0, s>>>(A, W, bias, C, M, N, K);
        else         gemm_bf16<0, 1><<<g, 256, 0, s>>>(A, W, bias, C, M, N, K);
    }
}

// ---------------- flash attention, one (b,h) x 128 q-rows per workgroup ----------------
// K/V tiles staged in LDS as f32 (V transposed to [dim][pos] during the copy);
// on gfx1250 staging uses GLOBAL_LOAD_ASYNC_TO_LDS (ASYNCcnt), else plain ld/st.
__global__ __launch_bounds__(256) void flash_attn(const float* __restrict__ qkv, float* __restrict__ ao) {
    const int bh = blockIdx.x, qblk = blockIdx.y;
    const int b = bh / H_, h = bh % H_;
    const int tid = threadIdx.x, wid = tid >> 5, lane = tid & 31;
    const int lanelo = lane & 15, hi = lane >> 4;
    const int qrow0 = qblk * 128 + wid * 16;
    const int RS = 3 * D_;                       // qkv row stride

    const float* qb   = qkv + (long)b * L_ * RS + h * HD_;
    const float* kptr = qb + D_;
    const float* vptr = qb + 2 * D_;

    __shared__ __align__(16) float Ksf[32 * 68];   // [pos][dim]   (pad 68: 16B-aligned rows)
    __shared__ __align__(16) float Vsf[64 * 34];   // [dim][pos]   (pad 34: 8B-aligned rows)
    __shared__ float Plds[8][16 * 32];
    float* P = Plds[wid];

    // Q fragments (scale 1/sqrt(HD)=0.125 folded in)
    v8u qf0, qf1;
    {
        const float* qr = qb + (long)(qrow0 + lanelo) * RS;
        #pragma unroll
        for (int p = 0; p < 8; ++p) {
            int k = kpat(p, hi);
            qf0[p] = pack2(qr[k] * 0.125f,  qr[k + 1] * 0.125f);
            qf1[p] = pack2(qr[k + 32] * 0.125f, qr[k + 33] * 0.125f);
        }
    }
    v16bf qa0 = __builtin_bit_cast(v16bf, qf0);
    v16bf qa1 = __builtin_bit_cast(v16bf, qf1);

    float mrow[8], srow[8];
    v8f O[4];
    #pragma unroll
    for (int r = 0; r < 8; ++r) { mrow[r] = -1e30f; srow[r] = 0.f; }
    #pragma unroll
    for (int t = 0; t < 4; ++t) O[t] = (v8f){0,0,0,0,0,0,0,0};

    for (int kb0 = 0; kb0 < L_; kb0 += 32) {
        // ---- cooperative K/V tile staging (32 pos x 64 dims each) ----
#ifdef ASYNC_LDS
        for (int i = tid; i < 2048; i += 256) {
            int pos = i >> 6, d = i & 63;
            async_cp_f32(kptr + (long)(kb0 + pos) * RS + d, &Ksf[pos * 68 + d]);
        }
        for (int i = tid; i < 2048; i += 256) {   // transpose during the async copy
            int pos = i >> 6, d = i & 63;
            async_cp_f32(vptr + (long)(kb0 + pos) * RS + d, &Vsf[d * 34 + pos]);
        }
        async_wait0();
#else
        for (int i = tid; i < 2048; i += 256) {
            int pos = i >> 6, d = i & 63;
            Ksf[pos * 68 + d] = kptr[(long)(kb0 + pos) * RS + d];
        }
        for (int i = tid; i < 2048; i += 256) {
            int pos = i >> 6, d = i & 63;
            Vsf[d * 34 + pos] = vptr[(long)(kb0 + pos) * RS + d];
        }
#endif
        __syncthreads();

        // ---- S = Q @ K^T for two 16-pos tiles ----
        v8f sarr[2];
        #pragma unroll
        for (int st = 0; st < 2; ++st) {
            const float* kr = &Ksf[(st * 16 + lanelo) * 68];
            v8u kf0, kf1;
            #pragma unroll
            for (int p = 0; p < 8; ++p) {
                int k = kpat(p, hi);
                float2 a = *(const float2*)(kr + k);
                float2 bvu = *(const float2*)(kr + 32 + k);
                kf0[p] = pack2(a.x, a.y);
                kf1[p] = pack2(bvu.x, bvu.y);
            }
            v8f sc = {0,0,0,0,0,0,0,0};
            sc = wmma_bf16(qa0, __builtin_bit_cast(v16bf, kf0), sc);
            sc = wmma_bf16(qa1, __builtin_bit_cast(v16bf, kf1), sc);
            sarr[st] = sc;
        }

        // ---- online softmax (lanes 0-15 / 16-31 hold disjoint row groups) ----
        #pragma unroll
        for (int r = 0; r < 8; ++r) {
            float cm = fmaxf(sarr[0][r], sarr[1][r]);
            cm = fmaxf(cm, __shfl_xor(cm, 1)); cm = fmaxf(cm, __shfl_xor(cm, 2));
            cm = fmaxf(cm, __shfl_xor(cm, 4)); cm = fmaxf(cm, __shfl_xor(cm, 8));
            float m2 = fmaxf(mrow[r], cm);
            float al = __expf(mrow[r] - m2);
            float p0v = __expf(sarr[0][r] - m2), p1v = __expf(sarr[1][r] - m2);
            float ls = p0v + p1v;
            ls += __shfl_xor(ls, 1); ls += __shfl_xor(ls, 2);
            ls += __shfl_xor(ls, 4); ls += __shfl_xor(ls, 8);
            srow[r] = srow[r] * al + ls;
            mrow[r] = m2;
            P[(r + 8 * hi) * 32 + lanelo]      = p0v;
            P[(r + 8 * hi) * 32 + 16 + lanelo] = p1v;
            O[0][r] *= al; O[1][r] *= al; O[2][r] *= al; O[3][r] *= al;
        }

        // ---- C-layout -> A-layout via per-wave LDS (DS ops in-order within a wave) ----
        v8u pf;
        #pragma unroll
        for (int p = 0; p < 8; ++p) {
            int cidx = kpat(p, hi);
            pf[p] = pack2(P[lanelo * 32 + cidx], P[lanelo * 32 + cidx + 1]);
        }
        v16bf pa = __builtin_bit_cast(v16bf, pf);

        // ---- O += P @ V ----
        #pragma unroll
        for (int t = 0; t < 4; ++t) {
            const float* vr = &Vsf[(t * 16 + lanelo) * 34];
            v8u vf;
            #pragma unroll
            for (int p = 0; p < 8; ++p) {
                float2 a = *(const float2*)(vr + kpat(p, hi));
                vf[p] = pack2(a.x, a.y);
            }
            O[t] = wmma_bf16(pa, __builtin_bit_cast(v16bf, vf), O[t]);
        }
        __syncthreads();   // protect Ksf/Vsf before next iteration's staging
    }

    #pragma unroll
    for (int r = 0; r < 8; ++r) {
        float inv = 1.f / srow[r];
        int row = qrow0 + r + 8 * hi;
        float* orow = ao + ((long)(b * L_ + row)) * D_ + h * HD_;
        #pragma unroll
        for (int t = 0; t < 4; ++t) orow[t * 16 + lanelo] = O[t][r] * inv;
    }
}

// ---------------- SSM sequential scan: h_t = h_{t-1} @ A^T + Bx_t  (one WG per batch) ----------------
__global__ __launch_bounds__(768) void ssm_scan(const float* __restrict__ Aexp,
                                                const float* __restrict__ Bx,
                                                float* __restrict__ hs) {
    const int b = blockIdx.x, d = threadIdx.x;
    __shared__ __align__(16) float h[D_];
    h[d] = 0.f;
    __syncthreads();
    const float4* Arow = (const float4*)(Aexp + (long)d * D_);
    for (int t = 0; t < L_; ++t) {
        float acc = Bx[(long)(b * L_ + t) * D_ + d];
        const float4* h4 = (const float4*)h;
        #pragma unroll 4
        for (int j = 0; j < D_ / 4; ++j) {
            float4 a = Arow[j]; float4 hv = h4[j];
            acc += a.x * hv.x + a.y * hv.y + a.z * hv.z + a.w * hv.w;
        }
        __syncthreads();
        h[d] = acc;
        hs[(long)(b * L_ + t) * D_ + d] = acc;
        __syncthreads();
    }
}

// ---------------- router probe conv (groups=8, K=3, pad=1) + relu + mean-pool ----------------
__global__ __launch_bounds__(256) void probe_pool(const float* __restrict__ x, const float* __restrict__ w,
                                                  const float* __restrict__ bias, float* __restrict__ pooled) {
    const int oc = blockIdx.x, b = blockIdx.y;
    const int ic0 = (oc / 24) * 96;
    __shared__ float ws[96 * 3];
    for (int i = threadIdx.x; i < 288; i += 256) ws[i] = w[oc * 288 + i];
    __syncthreads();
    float sum = 0.f;
    for (int l = threadIdx.x; l < L_; l += 256) {
        float a = bias[oc];
        #pragma unroll
        for (int k = 0; k < 3; ++k) {
            int lp = l + k - 1;
            if (lp >= 0 && lp < L_) {
                const float* xr = x + (long)(b * L_ + lp) * D_ + ic0;
                for (int ic = 0; ic < 96; ++ic) a += ws[ic * 3 + k] * xr[ic];
            }
        }
        sum += fmaxf(a, 0.f);
    }
    for (int off = 16; off; off >>= 1) sum += __shfl_xor(sum, off);
    __shared__ float red[8];
    if ((threadIdx.x & 31) == 0) red[threadIdx.x >> 5] = sum;
    __syncthreads();
    if (threadIdx.x == 0) {
        float s = 0.f;
        for (int i = 0; i < 8; ++i) s += red[i];
        pooled[b * 192 + oc] = s / (float)L_;
    }
}

// ---------------- router head: l1+relu, l2, gumbel-softmax ----------------
__global__ __launch_bounds__(128) void router_head(const float* __restrict__ pooled,
                                                   const float* __restrict__ w1, const float* __restrict__ b1,
                                                   const float* __restrict__ w2, const float* __restrict__ b2,
                                                   const float* __restrict__ temp, const float* __restrict__ gumbel,
                                                   float* __restrict__ rp) {
    __shared__ float hr[8 * 96];
    __shared__ float lg[16];
    const int t = threadIdx.x;
    for (int i = t; i < 768; i += 128) {
        int b = i / 96, j = i % 96;
        float a = b1[j];
        for (int k = 0; k < 192; ++k) a += pooled[b * 192 + k] * w1[k * 96 + j];
        hr[i] = fmaxf(a, 0.f);
    }
    __syncthreads();
    if (t < 16) {
        int b = t >> 1, cc = t & 1;
        float a = b2[cc];
        for (int k = 0; k < 96; ++k) a += hr[b * 96 + k] * w2[k * 2 + cc];
        lg[t] = a;
    }
    __syncthreads();
    if (t < 8) {
        float tau = fminf(fmaxf(temp[0], 0.1f), 10.f);
        float g0 = (lg[2 * t] + gumbel[2 * t]) / tau;
        float g1 = (lg[2 * t + 1] + gumbel[2 * t + 1]) / tau;
        float m = fmaxf(g0, g1);
        float e0 = __expf(g0 - m), e1 = __expf(g1 - m);
        rp[2 * t] = e0 / (e0 + e1); rp[2 * t + 1] = e1 / (e0 + e1);
    }
}

// ---------------- token gate scores (one wave per token) ----------------
__global__ __launch_bounds__(256) void token_scores(const float* __restrict__ x, const float* __restrict__ w,
                                                    const float* __restrict__ bias, float* __restrict__ sc) {
    const int wid = threadIdx.x >> 5, lane = threadIdx.x & 31;
    const int tok = blockIdx.x * 8 + wid;
    const float* xr = x + (long)tok * D_;
    float a = 0.f;
    for (int j = lane; j < D_; j += 32) a += xr[j] * w[j];
    for (int off = 16; off; off >>= 1) a += __shfl_xor(a, off);
    if (lane == 0) sc[tok] = tanhf(a + bias[0]);
}

// ---------------- top-k (K=512) mask by bisection on threshold ----------------
__global__ __launch_bounds__(1024) void topk_mask(const float* __restrict__ sc, float* __restrict__ mask) {
    const int b = blockIdx.x, t = threadIdx.x;
    const float s = sc[b * L_ + t];
    __shared__ int cnt;
    float lo = -1.01f, hi = 1.01f, mid = 0.f;
    for (int it = 0; it < 24; ++it) {
        mid = 0.5f * (lo + hi);
        if (t == 0) cnt = 0;
        __syncthreads();
        unsigned long long bal = __ballot(s > mid);
        if ((t & 31) == 0) atomicAdd(&cnt, __popcll(bal));
        __syncthreads();
        int c = cnt;
        if (c > 512) lo = mid; else hi = mid;
        __syncthreads();
    }
    mask[b * L_ + t] = (s >= mid) ? 1.f : 0.f;
}

// ---------------- ssm grouped conv (K=4, pad=3, truncated) fused with sigmoid-gate multiply ----------------
__global__ __launch_bounds__(256) void ssm_conv(const float* __restrict__ x, const float* __restrict__ w,
                                                const float* __restrict__ bias, const float* __restrict__ gate,
                                                float* __restrict__ out) {
    const int oc = blockIdx.x, b = blockIdx.y;
    const int ic0 = (oc / 96) * 96;
    __shared__ float ws[96 * 4];
    for (int i = threadIdx.x; i < 384; i += 256) ws[i] = w[oc * 384 + i];
    __syncthreads();
    for (int l = threadIdx.x; l < L_; l += 256) {
        float a = bias[oc];
        #pragma unroll
        for (int k = 0; k < 4; ++k) {
            int lp = l + k - 3;
            if (lp >= 0) {
                const float* xr = x + (long)(b * L_ + lp) * D_ + ic0;
                for (int ic = 0; ic < 96; ++ic) a += ws[ic * 4 + k] * xr[ic];
            }
        }
        long idx = (long)(b * L_ + l) * D_ + oc;
        out[idx] = a * gate[idx];
    }
}

// ---------------- layernorm (optional residual, optional relu) ----------------
__global__ __launch_bounds__(256) void layernorm_k(const float* __restrict__ in, const float* __restrict__ res,
                                                   const float* __restrict__ g, const float* __restrict__ bb,
                                                   float* __restrict__ out, int relu) {
    const long row = blockIdx.x;
    const float* ir = in + row * D_;
    const float* rr = res ? res + row * D_ : nullptr;
    float v[3], s = 0.f, s2 = 0.f;
    #pragma unroll
    for (int i = 0; i < 3; ++i) {
        int j = threadIdx.x + i * 256;
        float t = ir[j] + (rr ? rr[j] : 0.f);
        v[i] = t; s += t; s2 += t * t;
    }
    for (int off = 16; off; off >>= 1) { s += __shfl_xor(s, off); s2 += __shfl_xor(s2, off); }
    __shared__ float rs[8], rs2[8];
    __shared__ float mu, rstd;
    if ((threadIdx.x & 31) == 0) { rs[threadIdx.x >> 5] = s; rs2[threadIdx.x >> 5] = s2; }
    __syncthreads();
    if (threadIdx.x == 0) {
        float S = 0.f, S2 = 0.f;
        for (int i = 0; i < 8; ++i) { S += rs[i]; S2 += rs2[i]; }
        float m = S / (float)D_;
        mu = m; rstd = rsqrtf(S2 / (float)D_ - m * m + 1e-5f);
    }
    __syncthreads();
    #pragma unroll
    for (int i = 0; i < 3; ++i) {
        int j = threadIdx.x + i * 256;
        float t = (v[i] - mu) * rstd * g[j] + bb[j];
        if (relu) t = fmaxf(t, 0.f);
        out[row * D_ + j] = t;
    }
}

// ---------------- small elementwise kernels ----------------
__global__ void ew_exp(const float* in, float* out, int n) {
    int i = blockIdx.x * 256 + threadIdx.x;
    if (i < n) out[i] = __expf(in[i]);
}
__global__ void ew_axpyD(float* io, const float* x, const float* Dv, int n) {
    int i = blockIdx.x * 256 + threadIdx.x;
    if (i < n) io[i] += x[i] * Dv[i % D_];
}
__global__ void pos_build(float* pos) {
    int i = blockIdx.x * 256 + threadIdx.x;
    if (i < L_ * HD_) {
        int l = i >> 6, j = i & 63;
        float inv = __powf(10000.f, -(float)(j & 31) / 32.f);
        pos[i] = (j < 32) ? __sinf((float)l * inv) : __cosf((float)l * inv);
    }
}
__global__ void add_pos(float* qkv, const float* pos) {
    long i = (long)blockIdx.x * 256 + threadIdx.x;
    if (i < (long)B_ * L_ * 2 * D_) {
        int c = (int)(i % (2 * D_));
        long bl = i / (2 * D_);
        int l = (int)(bl % L_);
        qkv[bl * (3 * D_) + c] += pos[l * HD_ + (c & 63)];
    }
}
__global__ void fusion(const float* __restrict__ x, const float* __restrict__ ssm_feat,
                       const float* __restrict__ attn_feat, const float* __restrict__ ssm_enh,
                       const float* __restrict__ tr_enh, const float* __restrict__ gate,
                       const float* __restrict__ rp, const float* __restrict__ mask,
                       float* __restrict__ out, int n) {
    int i = blockIdx.x * 256 + threadIdx.x;
    if (i < n) {
        long bl = i / D_;
        int b = (int)(bl / L_), l = (int)(bl % L_);
        float gt = gate[i];
        float so = ssm_feat[i] + gt * tr_enh[i];
        float to = attn_feat[i] + (1.f - gt) * ssm_enh[i];
        float mix = rp[2 * b] * so + rp[2 * b + 1] * to;
        float m = mask[b * L_ + l];
        out[i] = m * mix + (1.f - m) * x[i];
    }
}

// ---------------- orchestration ----------------
extern "C" void kernel_launch(void* const* d_in, const int* in_sizes, int n_in,
                              void* d_out, int out_size, void* d_ws, size_t ws_size,
                              hipStream_t stream) {
    const float* x     = (const float*)d_in[0];
    const float* gum   = (const float*)d_in[1];
    const float* pc_w  = (const float*)d_in[2];   const float* pc_b  = (const float*)d_in[3];
    const float* l1_w  = (const float*)d_in[4];   const float* l1_b  = (const float*)d_in[5];
    const float* l2_w  = (const float*)d_in[6];   const float* l2_b  = (const float*)d_in[7];
    const float* temp  = (const float*)d_in[8];
    const float* tg_w  = (const float*)d_in[9];   const float* tg_b  = (const float*)d_in[10];
    const float* logA  = (const float*)d_in[11];
    const float* B_w   = (const float*)d_in[12];  const float* B_b   = (const float*)d_in[13];
    const float* C_w   = (const float*)d_in[14];  const float* C_b   = (const float*)d_in[15];
    const float* Dvec  = (const float*)d_in[16];
    const float* sc_w  = (const float*)d_in[17];  const float* sc_b  = (const float*)d_in[18];
    const float* cg_w  = (const float*)d_in[19];  const float* cg_b  = (const float*)d_in[20];
    const float* so_w  = (const float*)d_in[21];  const float* so_b  = (const float*)d_in[22];
    const float* sn_g  = (const float*)d_in[23];  const float* sn_b  = (const float*)d_in[24];
    const float* qkv_w = (const float*)d_in[25];
    const float* ao_w  = (const float*)d_in[26];  const float* ao_b  = (const float*)d_in[27];
    const float* s2t_w1 = (const float*)d_in[28]; const float* s2t_b1 = (const float*)d_in[29];
    const float* s2t_g  = (const float*)d_in[30]; const float* s2t_bn = (const float*)d_in[31];
    const float* s2t_w2 = (const float*)d_in[32]; const float* s2t_b2 = (const float*)d_in[33];
    const float* t2s_w1 = (const float*)d_in[34]; const float* t2s_b1 = (const float*)d_in[35];
    const float* t2s_g  = (const float*)d_in[36]; const float* t2s_bn = (const float*)d_in[37];
    const float* t2s_w2 = (const float*)d_in[38]; const float* t2s_b2 = (const float*)d_in[39];
    const float* g_w    = (const float*)d_in[40]; const float* g_b    = (const float*)d_in[41];
    float* out = (float*)d_out;

    const long BLD = (long)B_ * L_ * D_;   // 6291456
    float* w      = (float*)d_ws;
    float* Aexp   = w;                    // 589824
    float* pos    = Aexp + 589824;        // 65536
    float* pooled = pos + 65536;          // 1536
    float* rp     = pooled + 1536;        // 16
    float* sc     = rp + 16;              // 8192
    float* mask   = sc + 8192;            // 8192
    float* buf1   = mask + 8192;
    float* buf2   = buf1 + BLD;
    float* buf3   = buf2 + BLD;
    float* qkvb   = buf3 + BLD;           // B*L*3D
    float* enh1   = qkvb;                 // reuse qkv region after attention
    float* enh2   = qkvb + BLD;

    const int M = B_ * L_;                // 8192

    // ---- router ----
    ew_exp<<<(589824 + 255) / 256, 256, 0, stream>>>(logA, Aexp, 589824);
    pos_build<<<(L_ * HD_ + 255) / 256, 256, 0, stream>>>(pos);
    probe_pool<<<dim3(192, B_), 256, 0, stream>>>(x, pc_w, pc_b, pooled);
    router_head<<<1, 128, 0, stream>>>(pooled, l1_w, l1_b, l2_w, l2_b, temp, gum, rp);
    token_scores<<<M / 8, 256, 0, stream>>>(x, tg_w, tg_b, sc);
    topk_mask<<<B_, 1024, 0, stream>>>(sc, mask);

    // ---- SSM branch ----
    launch_gemm(x, B_w, B_b, buf1, M, D_, D_, 0, 0, stream);          // Bx
    ssm_scan<<<B_, D_, 0, stream>>>(Aexp, buf1, buf2);                // hs
    launch_gemm(buf2, C_w, C_b, buf1, M, D_, D_, 0, 0, stream);       // C(hs)
    ew_axpyD<<<(int)((BLD + 255) / 256), 256, 0, stream>>>(buf1, x, Dvec, (int)BLD);
    launch_gemm(x, cg_w, cg_b, buf2, M, D_, D_, 2, 0, stream);        // sigmoid conv gate
    ssm_conv<<<dim3(D_, B_), 256, 0, stream>>>(x, sc_w, sc_b, buf2, buf3);
    launch_gemm(buf1, so_w, so_b, buf2, M, D_, D_, 0, 0, stream);     // ssm half
    launch_gemm(buf3, so_w + D_ * D_, nullptr, buf2, M, D_, D_, 0, 1, stream); // conv half
    layernorm_k<<<M, 256, 0, stream>>>(buf2, x, sn_g, sn_b, buf1, 0); // ssm_feat -> buf1

    // ---- attention branch ----
    launch_gemm(x, qkv_w, nullptr, qkvb, M, 3 * D_, D_, 0, 0, stream);
    add_pos<<<(int)(((long)M * 2 * D_ + 255) / 256), 256, 0, stream>>>(qkvb, pos);
    flash_attn<<<dim3(B_ * H_, L_ / 128), 256, 0, stream>>>(qkvb, buf2);   // ao -> buf2
    launch_gemm(buf2, ao_w, ao_b, buf3, M, D_, D_, 0, 0, stream);          // attn_feat -> buf3

    // ---- bridge ----
    launch_gemm(buf1, s2t_w1, s2t_b1, buf2, M, D_, D_, 0, 0, stream);
    layernorm_k<<<M, 256, 0, stream>>>(buf2, nullptr, s2t_g, s2t_bn, buf2, 1);
    launch_gemm(buf2, s2t_w2, s2t_b2, enh1, M, D_, D_, 0, 0, stream);      // ssm_enh
    launch_gemm(buf3, t2s_w1, t2s_b1, buf2, M, D_, D_, 0, 0, stream);
    layernorm_k<<<M, 256, 0, stream>>>(buf2, nullptr, t2s_g, t2s_bn, buf2, 1);
    launch_gemm(buf2, t2s_w2, t2s_b2, enh2, M, D_, D_, 0, 0, stream);      // tr_enh
    launch_gemm(enh1, g_w, nullptr, buf2, M, D_, D_, 0, 0, stream);
    launch_gemm(enh2, g_w + D_ * D_, g_b, buf2, M, D_, D_, 2, 1, stream);  // sigmoid gate

    // ---- fusion ----
    fusion<<<(int)((BLD + 255) / 256), 256, 0, stream>>>(x, buf1, buf3, enh1, enh2, buf2,
                                                         rp, mask, out, (int)BLD);
}
